// Attention_74577812128172
// MI455X (gfx1250) — matrix operand order repaired and checked by reading, not compiled, not run on hardware
//
#include <hip/hip_runtime.h>
#include <math.h>

typedef __attribute__((ext_vector_type(16))) __bf16 v16bf;
typedef __attribute__((ext_vector_type(8)))  __bf16 v8bf;
typedef __attribute__((ext_vector_type(8)))  float  v8f;
typedef __attribute__((ext_vector_type(4)))  unsigned int u32x4;
typedef __attribute__((ext_vector_type(4)))  int i32x4;
typedef __attribute__((ext_vector_type(8)))  int i32x8;

namespace {

constexpr int kB = 2, kH = 16, kS = 2048, kF = 128;
constexpr float kScale = 0.08838834764831845f; // 1/sqrt(128)

// ---- WMMA fragment loaders (wave32, 16-bit operands) -----------------------
// A matrix 16x32 (M x K): lane<16 -> M=lane, K elems k0..k0+15
//                         lane>=16 -> M=lane-16, K elems k0+16..k0+31
__device__ __forceinline__ v16bf load_afrag(const __bf16* base, int stride,
                                            int row0, int k0, int lane) {
  const int lr = lane & 15, half = lane >> 4;
  const __bf16* p = base + (row0 + lr) * stride + k0 + half * 16;
  v16bf a;
#pragma unroll
  for (int i = 0; i < 8; ++i) { a[i] = p[i]; a[8 + i] = p[8 + i]; }
  return a;
}

// Vectorized variant for global memory: two 16-byte loads per fragment.
__device__ __forceinline__ v16bf load_afrag_vec(const __bf16* base, int stride,
                                                int row0, int k0, int lane) {
  const int lr = lane & 15, half = lane >> 4;
  const __bf16* p = base + (size_t)(row0 + lr) * stride + k0 + half * 16;
  v8bf lo = *(const v8bf*)p;
  v8bf hi = *(const v8bf*)(p + 8);
  v16bf a;
#pragma unroll
  for (int i = 0; i < 8; ++i) { a[i] = lo[i]; a[8 + i] = hi[i]; }
  return a;
}

// B matrix 32x16 (K x N): lane<16 -> N=lane, K = k0+0..15 ; lane>=16 -> N=lane-16, K = k0+16..31
// Memory holds src[n][k] (K contiguous per n-row), e.g. W[e][f] or Kmat[key][f].
__device__ __forceinline__ v16bf load_bfrag_rowmajorK(const __bf16* base, int stride,
                                                      int n0, int k0, int lane) {
  const int lr = lane & 15, half = lane >> 4;
  const __bf16* p = base + (n0 + lr) * stride + k0 + half * 16;
  v16bf b;
#pragma unroll
  for (int i = 0; i < 16; ++i) b[i] = p[i];
  return b;
}

// Memory holds src[k][n] (row-major in k), e.g. V[key][e] -> strided gather over k.
__device__ __forceinline__ v16bf load_bfrag_colmajorK(const __bf16* base, int stride,
                                                      int n0, int k0, int lane) {
  const int lr = lane & 15, half = lane >> 4;
  const __bf16* p = base + (k0 + half * 16) * stride + n0 + lr;
  v16bf b;
#pragma unroll
  for (int i = 0; i < 16; ++i) b[i] = p[i * stride];
  return b;
}

__device__ __forceinline__ v8f zero_v8f() {
  v8f z;
#pragma unroll
  for (int j = 0; j < 8; ++j) z[j] = 0.0f;
  return z;
}

// Low 32 bits of the flat address of an LDS object == LDS byte offset
// (flat LDS aperture: addr[63:32] selects aperture, addr[31:0] is the offset).
__device__ __forceinline__ unsigned lds_addr_of(const void* p) {
  return (unsigned)(unsigned long long)p;
}

// ---- Tensor Data Mover: DMA one contiguous 32KB (16384 x 2B) tile to LDS ----
// D# group0: count=1 | lds_addr | global_addr[31:0] | global_addr[56:32], type=2
// D# group1: data_size=2B, tensor_dim0=tile_dim0=16384 (1-D tile), dims 1/2 unused
__device__ __forceinline__ void tdm_load_32kb(const void* gsrc, unsigned lds_byte_off) {
  const unsigned long long ga = (unsigned long long)gsrc;
  u32x4 g0;
  g0[0] = 1u;                                   // count=1, user descriptor
  g0[1] = lds_byte_off;                         // lds_addr
  g0[2] = (unsigned)ga;                         // global_addr[31:0]
  g0[3] = ((unsigned)(ga >> 32) & 0x01FFFFFFu)  // global_addr[56:32]
          | 0x80000000u;                        // type=2 ("image")
  i32x8 g1;
  g1[0] = 0x00010000;  // data_size=1 (2 bytes); workgroup_mask=0
  g1[1] = 0x40000000;  // tensor_dim0[15:0]=16384 in bits[63:48]
  g1[2] = 0x00010000;  // tensor_dim1[15:0]=1 in bits[95:80]; tensor_dim0[31:16]=0
  g1[3] = 0x40000000;  // tile_dim0=16384 in bits[127:112]; tensor_dim1[31:16]=0
  g1[4] = 0;           // tile_dim1=0 (unused), tile_dim2=0
  g1[5] = 16384;       // tensor_dim0_stride[31:0]
  g1[6] = 0;
  g1[7] = 0;
  i32x4 z4 = {0, 0, 0, 0};
#if defined(__clang_major__) && (__clang_major__ >= 23)
  i32x8 z8 = {0, 0, 0, 0, 0, 0, 0, 0};
  __builtin_amdgcn_tensor_load_to_lds(g0, g1, z4, z4, z8, 0);
#else
  __builtin_amdgcn_tensor_load_to_lds(g0, g1, z4, z4, 0);
#endif
}

} // namespace

// ---------------------------------------------------------------------------
// Kernel 1: per-head QKV projection, fp32 -> bf16, y = x @ W^T + b
// grid = B*H*(S/128) = 512 blocks, 256 threads (8 waves)
// ---------------------------------------------------------------------------
__global__ __launch_bounds__(256) void qkv_proj_kernel(
    const float* __restrict__ x,
    const float* __restrict__ Wq, const float* __restrict__ Wk, const float* __restrict__ Wv,
    const float* __restrict__ bq, const float* __restrict__ bk, const float* __restrict__ bv,
    __bf16* __restrict__ Qw, __bf16* __restrict__ Kw, __bf16* __restrict__ Vw) {
  __shared__ __bf16 Xs[128 * kF];  // 32 KB
  __shared__ __bf16 Ws[128 * kF];  // 32 KB  (64 KB total static LDS)

  const int tid = threadIdx.x, lane = tid & 31, wid = tid >> 5;
  const int bh = blockIdx.x >> 4;   // b*16 + h
  const int st = blockIdx.x & 15;   // 128-row s tile
  const int h = bh & 15;
  const int lr = lane & 15, half = lane >> 4;

  const float* xb = x + ((size_t)bh * kS + (size_t)st * 128) * kF;
#pragma unroll 4
  for (int i = 0; i < 64; ++i) {
    int idx = i * 256 + tid;
    int r = idx >> 7, c = idx & 127;
    Xs[r * kF + c] = (__bf16)xb[(size_t)r * kF + c];
  }
  __syncthreads();

  // A fragments for this wave's 16-row strip (reused for Q, K and V weights)
  v16bf af[4];
#pragma unroll
  for (int kc = 0; kc < 4; ++kc) af[kc] = load_afrag(Xs, kF, wid * 16, kc * 32, lane);

  const float* Wm[3] = {Wq + (size_t)h * kF * kF, Wk + (size_t)h * kF * kF, Wv + (size_t)h * kF * kF};
  const float* bm[3] = {bq + (size_t)h * kF, bk + (size_t)h * kF, bv + (size_t)h * kF};
  __bf16* Om[3] = {Qw, Kw, Vw};

  for (int m = 0; m < 3; ++m) {
    __syncthreads();  // previous Ws fully consumed
#pragma unroll 4
    for (int i = 0; i < 64; ++i) {
      int idx = i * 256 + tid;
      int r = idx >> 7, c = idx & 127;
      Ws[r * kF + c] = (__bf16)Wm[m][(size_t)r * kF + c];
    }
    __syncthreads();

    v8f acc[8];
#pragma unroll
    for (int nt = 0; nt < 8; ++nt) acc[nt] = zero_v8f();

#pragma unroll
    for (int kc = 0; kc < 4; ++kc) {
#pragma unroll
      for (int nt = 0; nt < 8; ++nt) {
        v16bf bf = load_bfrag_rowmajorK(Ws, kF, nt * 16, kc * 32, lane);
        acc[nt] = __builtin_amdgcn_wmma_f32_16x16x32_bf16(
            false, af[kc], false, bf, (short)0, acc[nt], false, false);
      }
    }

    // bias add + store bf16 in [S][F] row-major per (b,h)
    __bf16* ob = Om[m] + ((size_t)bh * kS + (size_t)st * 128 + (size_t)wid * 16) * kF;
#pragma unroll
    for (int nt = 0; nt < 8; ++nt) {
      float bvv = bm[m][nt * 16 + lr];
#pragma unroll
      for (int j = 0; j < 8; ++j) {
        int r = j + half * 8;                 // C/D layout: vgpr j -> row j (+8 for hi half)
        ob[(size_t)r * kF + nt * 16 + lr] = (__bf16)(acc[nt][j] + bvv);
      }
    }
  }
}

// ---------------------------------------------------------------------------
// Kernel 2: causal flash attention, bf16 WMMA, TDM double-buffered K/V tiles
// grid = B*H*(S/128) = 512 blocks, 256 threads (8 waves); each wave: 16 q-rows
// dynamic LDS: 2 x (32KB K + 32KB V) = 128 KB
// LDS map (element offsets in smem): K tiles at {0, 16384}+koff toggle,
//                                    V tiles at 32768 + {0, 16384}
// Buffer selection is done with INTEGER offsets only, so all tile accesses
// stay in address-space 3 and lower to ds_load_b128 (no flat ops).
// ---------------------------------------------------------------------------
__global__ __launch_bounds__(256) void flash_attn_kernel(
    const __bf16* __restrict__ Qw, const __bf16* __restrict__ Kw,
    const __bf16* __restrict__ Vw, float* __restrict__ out) {
  extern __shared__ __align__(16) __bf16 smem[];

  const int tid = threadIdx.x, lane = tid & 31, wid = tid >> 5;
  const int lr = lane & 15, half = lane >> 4;
  const int bh = blockIdx.x >> 4;
  const int qb = blockIdx.x & 15;
  const size_t base = (size_t)bh * kS * kF;
  const int q0 = qb * 128 + wid * 16;   // first query row of this wave

  // kick off first K/V tile DMA (wave0 -> K, wave1 -> V; TDM is wave-level)
  if (wid == 0)      tdm_load_32kb(Kw + base, lds_addr_of(&smem[0]));
  else if (wid == 1) tdm_load_32kb(Vw + base, lds_addr_of(&smem[32768]));

  // Q fragments straight from global (bf16, row-major [S][F]) while DMA runs
  v16bf qf[4];
#pragma unroll
  for (int kc = 0; kc < 4; ++kc)
    qf[kc] = load_afrag_vec(Qw + base, kF, q0, kc * 32, lane);

  float mrun[8], lrun[8];
  v8f oacc[8];
#pragma unroll
  for (int j = 0; j < 8; ++j) { mrun[j] = -__builtin_inff(); lrun[j] = 0.0f; }
#pragma unroll
  for (int nt = 0; nt < 8; ++nt) oacc[nt] = zero_v8f();

  if (wid < 2) __builtin_amdgcn_s_wait_tensorcnt(0);
  __syncthreads();  // first tiles ready

  int koff = 0;     // element offset of current K tile; V tile at 32768+koff
  for (int kb = 0; kb <= qb; ++kb) {
    const int knext = koff ^ 16384;
    // prefetch next K/V tiles into the other buffers while we compute
    if (kb < qb) {
      const size_t nb = base + (size_t)(kb + 1) * 128 * kF;
      if (wid == 0)      tdm_load_32kb(Kw + nb, lds_addr_of(&smem[knext]));
      else if (wid == 1) tdm_load_32kb(Vw + nb, lds_addr_of(&smem[32768 + knext]));
    }

    // ---- S = Q @ K^T for this 16x128 strip ----
    v8f sacc[8];
#pragma unroll
    for (int nt = 0; nt < 8; ++nt) sacc[nt] = zero_v8f();
#pragma unroll
    for (int kc = 0; kc < 4; ++kc) {
#pragma unroll
      for (int nt = 0; nt < 8; ++nt) {
        v16bf kf = load_bfrag_rowmajorK(smem + koff, kF, nt * 16, kc * 32, lane);
        sacc[nt] = __builtin_amdgcn_wmma_f32_16x16x32_bf16(
            false, qf[kc], false, kf, (short)0, sacc[nt], false, false);
      }
    }
    __syncthreads();  // all waves done reading K[cur] before P overwrites it

    const int poff = koff + wid * (16 * kF);  // per-wave 16x128 P tile aliases K[cur]

    // ---- mask + scale + online softmax ----
    const bool diag = (kb == qb);
    float mloc[8];
#pragma unroll
    for (int j = 0; j < 8; ++j) mloc[j] = -__builtin_inff();
#pragma unroll
    for (int nt = 0; nt < 8; ++nt) {
      const int key = kb * 128 + nt * 16 + lr;
#pragma unroll
      for (int j = 0; j < 8; ++j) {
        float s = sacc[nt][j] * kScale;
        const int q = q0 + j + half * 8;
        if (diag && key > q) s = -__builtin_inff();
        sacc[nt][j] = s;
        mloc[j] = fmaxf(mloc[j], s);
      }
    }
    // half-wave (16 lane) reductions; xor masks < 16 keep both halves separate
#pragma unroll
    for (int off = 1; off < 16; off <<= 1)
#pragma unroll
      for (int j = 0; j < 8; ++j) mloc[j] = fmaxf(mloc[j], __shfl_xor(mloc[j], off, 32));

    float alpha[8];
#pragma unroll
    for (int j = 0; j < 8; ++j) {
      float mn = fmaxf(mrun[j], mloc[j]);
      alpha[j] = __expf(mrun[j] - mn);
      mrun[j] = mn;
    }

    float lloc[8];
#pragma unroll
    for (int j = 0; j < 8; ++j) lloc[j] = 0.0f;
#pragma unroll
    for (int nt = 0; nt < 8; ++nt) {
#pragma unroll
      for (int j = 0; j < 8; ++j) {
        float p = __expf(sacc[nt][j] - mrun[j]);
        lloc[j] += p;
        smem[poff + (j + half * 8) * kF + nt * 16 + lr] = (__bf16)p;  // C/D -> row-major P
      }
    }
#pragma unroll
    for (int off = 1; off < 16; off <<= 1)
#pragma unroll
      for (int j = 0; j < 8; ++j) lloc[j] += __shfl_xor(lloc[j], off, 32);
#pragma unroll
    for (int j = 0; j < 8; ++j) lrun[j] = lrun[j] * alpha[j] + lloc[j];

    // rescale running O (row j of C/D layout matches alpha[j])
#pragma unroll
    for (int nt = 0; nt < 8; ++nt)
#pragma unroll
      for (int j = 0; j < 8; ++j) oacc[nt][j] *= alpha[j];

    // ---- O += P @ V (same-wave LDS ops are in order: P store -> load ok) ----
#pragma unroll
    for (int kc = 0; kc < 4; ++kc) {
      v16bf pf = load_afrag(smem + poff, kF, 0, kc * 32, lane);
#pragma unroll
      for (int nt = 0; nt < 8; ++nt) {
        v16bf vf = load_bfrag_colmajorK(smem + 32768 + koff, kF, nt * 16, kc * 32, lane);
        oacc[nt] = __builtin_amdgcn_wmma_f32_16x16x32_bf16(
            false, pf, false, vf, (short)0, oacc[nt], false, false);
      }
    }

    // issuing waves wait for next-tile DMA; barrier fences this iter's LDS
    // reads (incl. P) before the next prefetch targets the retired buffer
    if (kb < qb && wid < 2) __builtin_amdgcn_s_wait_tensorcnt(0);
    __syncthreads();
    koff = knext;
  }

  // ---- finalize: out = O / l (fp32) ----
  float* ob = out + base + (size_t)q0 * kF;
#pragma unroll
  for (int nt = 0; nt < 8; ++nt) {
#pragma unroll
    for (int j = 0; j < 8; ++j) {
      ob[(size_t)(j + half * 8) * kF + nt * 16 + lr] = oacc[nt][j] / lrun[j];
    }
  }
}

// ---------------------------------------------------------------------------
extern "C" void kernel_launch(void* const* d_in, const int* in_sizes, int n_in,
                              void* d_out, int out_size, void* d_ws, size_t ws_size,
                              hipStream_t stream) {
  (void)in_sizes; (void)n_in; (void)out_size; (void)ws_size;
  const float* x  = (const float*)d_in[0];
  const float* Wq = (const float*)d_in[1];
  const float* Wk = (const float*)d_in[2];
  const float* Wv = (const float*)d_in[3];
  const float* bq = (const float*)d_in[4];
  const float* bk = (const float*)d_in[5];
  const float* bv = (const float*)d_in[6];
  float* out = (float*)d_out;

  const size_t qkvElems = (size_t)kB * kH * kS * kF;  // 8,388,608 bf16 each
  __bf16* Qw = (__bf16*)d_ws;
  __bf16* Kw = Qw + qkvElems;
  __bf16* Vw = Kw + qkvElems;

  const int nblocks = kB * kH * (kS / 128);  // 512
  qkv_proj_kernel<<<nblocks, 256, 0, stream>>>(x, Wq, Wk, Wv, bq, bk, bv, Qw, Kw, Vw);
  flash_attn_kernel<<<nblocks, 256, 131072, stream>>>(Qw, Kw, Vw, out);
}
